// EasyPeasyConvRelationalNet_18657337934451
// MI455X (gfx1250) — matrix operand order repaired and checked by reading, not compiled, not run on hardware
//
#include <hip/hip_runtime.h>

typedef __attribute__((ext_vector_type(16))) _Float16 v16h;
typedef __attribute__((ext_vector_type(8)))  _Float16 v8h;
typedef __attribute__((ext_vector_type(8)))  float    v8f;

namespace {
constexpr int kB = 32, kC = 64, kL = 144, kD = 65, kF = 128, kOut = 64;
constexpr float kSlope = 0.01f;
constexpr int kLdsStride = kF + 8;   // 136 halves -> 272B rows, keeps b128 alignment
}

__device__ __forceinline__ float lrelu(float v) { return v > 0.f ? v : kSlope * v; }

// ---------------------------------------------------------------------------
// Kernel 1: per-node projections through the (factorized) first g layer.
//   Aproj[b,m,f] = nodes[b,m,:] @ g_w0[0:65,  f]
//   Bproj[b,l,f] = nodes[b,l,:] @ g_w0[65:130,f] + g_b0[f]
// nodes[b,p,c] = x_img[b,c,p] for c<64, nodes[b,p,64] = p
// ---------------------------------------------------------------------------
__global__ void k_proj(const float* __restrict__ x_img,
                       const float* __restrict__ g_w0,
                       const float* __restrict__ g_b0,
                       float* __restrict__ Aproj,
                       float* __restrict__ Bproj) {
  int tid = blockIdx.x * blockDim.x + threadIdx.x;
  if (tid >= kB * kL * kF) return;
  int f = tid & (kF - 1);
  int m = (tid >> 7) % kL;
  int b = tid / (kL * kF);
  const float* wi = g_w0 + f;              // rows 0..64   (pairs with nodes[b,m])
  const float* wj = g_w0 + kD * kF + f;    // rows 65..129 (pairs with nodes[b,l])
  const float* img = x_img + (size_t)(b * kC) * kL + m;
  float accA = 0.f, accB = 0.f;
#pragma unroll 4
  for (int d = 0; d < kC; ++d) {
    float nd = img[(size_t)d * kL];
    accA += nd * wi[d * kF];
    accB += nd * wj[d * kF];
  }
  float coord = (float)m;                  // appended coordinate feature
  accA += coord * wi[kC * kF];
  accB += coord * wj[kC * kF];
  accB += g_b0[f];                         // fold layer-1 bias into Bproj
  Aproj[tid] = accA;
  Bproj[tid] = accB;
}

// ---------------------------------------------------------------------------
// Kernel 2: pre-swizzle g_w1 (f32 row-major [k][n]) into per-lane WMMA
// B-fragment order. Fragment (kk, wave, lane) holds 16 f16: element e is
// g_w1[32*kk + 8*(lane>=16) + (e&7) + 16*(e>=8)][16*wave + (lane&15)].
// ---------------------------------------------------------------------------
__global__ void k_pack_w1(const float* __restrict__ g_w1,
                          _Float16* __restrict__ w1frag) {
  int t = blockIdx.x * blockDim.x + threadIdx.x;   // 0 .. 16383
  int e    = t & 15;
  int lane = (t >> 4) & 31;
  int w    = (t >> 9) & 7;
  int kk   = t >> 12;
  int k = kk * 32 + ((lane >> 4) << 3) + (e & 7) + ((e >> 3) << 4);
  int n = (w << 4) + (lane & 15);
  w1frag[t] = (_Float16)g_w1[k * kF + n];
}

// ---------------------------------------------------------------------------
// Kernel 3: one workgroup per (b,l). Build h1 = lrelu(Aproj[b,m,:]+Bproj[b,l,:])
// in LDS (f16), GEMM against g_w1 with v_wmma_f32_16x16x32_f16 (bias folded
// into acc init), lrelu, column-sum over m in registers, write 128 partials.
// ---------------------------------------------------------------------------
__global__ __launch_bounds__(256) void k_pair(
    const float* __restrict__ Aproj, const float* __restrict__ Bproj,
    const _Float16* __restrict__ w1frag, const float* __restrict__ g_b1,
    float* __restrict__ part) {
  __shared__ _Float16 sh[kL * kLdsStride];   // 39168 B
  int b = blockIdx.x / kL;
  int l = blockIdx.x % kL;
  const float* Ab = Aproj + (size_t)b * kL * kF;
  const float* Bl = Bproj + ((size_t)b * kL + l) * kF;

  // Stage h1 tile (144 x 128 f16) row-major in LDS.
  for (int e = threadIdx.x; e < kL * kF; e += 256) {
    int m = e >> 7, k = e & (kF - 1);
    sh[m * kLdsStride + k] = (_Float16)lrelu(Ab[e] + Bl[k]);
  }
  __syncthreads();

  int wave = threadIdx.x >> 5, lane = threadIdx.x & 31;
  int n0 = wave << 4, ln = lane & 15;
  int kb = (lane >> 4) << 3;                 // A-frag K base: lanes 16-31 -> +8

  v16h bfrag[4];                             // g_w1 fragments, one per K-step
#pragma unroll
  for (int kk = 0; kk < 4; ++kk)
    bfrag[kk] = *(const v16h*)(w1frag + (((kk * 8 + wave) * 32 + lane) << 4));

  float bias = g_b1[n0 + ln];                // layer-2 bias folded into C init
  float colsum = 0.f;
  for (int mt = 0; mt < 9; ++mt) {           // 144 rows = 9 M-tiles
    const _Float16* row = sh + (mt * 16 + ln) * kLdsStride + kb;
    v8f acc = {bias, bias, bias, bias, bias, bias, bias, bias};
#pragma unroll
    for (int kk = 0; kk < 4; ++kk) {
      v8h lo = *(const v8h*)(row + kk * 32);        // K = kb .. kb+7
      v8h hi = *(const v8h*)(row + kk * 32 + 16);   // K = kb+16 .. kb+23
      v16h a = __builtin_shufflevector(lo, hi, 0, 1, 2, 3, 4, 5, 6, 7,
                                       8, 9, 10, 11, 12, 13, 14, 15);
      acc = __builtin_amdgcn_wmma_f32_16x16x32_f16(
          false, a, false, bfrag[kk], (short)0, acc, false, false);
    }
#pragma unroll
    for (int r = 0; r < 8; ++r) colsum += lrelu(acc[r]);
  }
  // combine M halves (lanes 0-15 hold M%16<8, lanes 16-31 hold M%16>=8)
  colsum += __shfl_xor(colsum, 16, 32);
  if (lane < 16) part[(size_t)blockIdx.x * kF + n0 + lane] = colsum;
}

// ---------------------------------------------------------------------------
// Kernel 4: reduce partials over l, take mean over L^2, run the tiny post MLP
// and output layer. One workgroup; everything LDS-resident; deterministic.
// ---------------------------------------------------------------------------
__global__ __launch_bounds__(256) void k_post(
    const float* __restrict__ part,
    const float* __restrict__ p_w0, const float* __restrict__ p_b0,
    const float* __restrict__ p_w1, const float* __restrict__ p_b1,
    const float* __restrict__ o_w,  const float* __restrict__ o_b,
    float* __restrict__ out) {
  __shared__ float buf0[kB * kF];
  __shared__ float buf1[kB * kF];
  const float inv = 1.f / (float)(kL * kL);
  for (int i = threadIdx.x; i < kB * kF; i += 256) {
    int b = i >> 7, f = i & (kF - 1);
    float s = 0.f;
    const float* p = part + (size_t)b * kL * kF + f;
    for (int l = 0; l < kL; ++l) s += p[l * kF];
    buf0[i] = s * inv;
  }
  __syncthreads();
  for (int i = threadIdx.x; i < kB * kF; i += 256) {
    int b = i >> 7, f = i & (kF - 1);
    float acc = p_b0[f];
    for (int k = 0; k < kF; ++k) acc += buf0[(b << 7) + k] * p_w0[k * kF + f];
    buf1[i] = lrelu(acc);
  }
  __syncthreads();
  for (int i = threadIdx.x; i < kB * kF; i += 256) {
    int b = i >> 7, f = i & (kF - 1);
    float acc = p_b1[f];
    for (int k = 0; k < kF; ++k) acc += buf1[(b << 7) + k] * p_w1[k * kF + f];
    buf0[i] = lrelu(acc);
  }
  __syncthreads();
  for (int i = threadIdx.x; i < kB * kOut; i += 256) {
    int b = i >> 6, o = i & (kOut - 1);
    float acc = o_b[o];
    for (int k = 0; k < kF; ++k) acc += buf0[(b << 7) + k] * o_w[k * kOut + o];
    out[i] = lrelu(acc);
  }
}

// ---------------------------------------------------------------------------
extern "C" void kernel_launch(void* const* d_in, const int* in_sizes, int n_in,
                              void* d_out, int out_size, void* d_ws, size_t ws_size,
                              hipStream_t stream) {
  (void)in_sizes; (void)n_in; (void)out_size; (void)ws_size;
  const float* x_img = (const float*)d_in[0];
  const float* g_w0  = (const float*)d_in[1];
  const float* g_b0  = (const float*)d_in[2];
  const float* g_w1  = (const float*)d_in[3];
  const float* g_b1  = (const float*)d_in[4];
  const float* p_w0  = (const float*)d_in[5];
  const float* p_b0  = (const float*)d_in[6];
  const float* p_w1  = (const float*)d_in[7];
  const float* p_b1  = (const float*)d_in[8];
  const float* o_w   = (const float*)d_in[9];
  const float* o_b   = (const float*)d_in[10];

  const size_t nBLF = (size_t)kB * kL * kF;        // 589824
  float*    Aproj  = (float*)d_ws;
  float*    Bproj  = Aproj + nBLF;
  float*    part   = Bproj + nBLF;                 // per-(b,l) column sums
  _Float16* w1frag = (_Float16*)(part + nBLF);     // 16384 halves

  k_proj  <<<(int)((nBLF + 255) / 256), 256, 0, stream>>>(x_img, g_w0, g_b0, Aproj, Bproj);
  k_pack_w1<<<64, 256, 0, stream>>>(g_w1, w1frag);
  k_pair  <<<kB * kL, 256, 0, stream>>>(Aproj, Bproj, w1frag, g_b1, part);
  k_post  <<<1, 256, 0, stream>>>(part, p_w0, p_b0, p_w1, p_b1, o_w, o_b, (float*)d_out);
}
